// CrossAttention_9698036154741
// MI455X (gfx1250) — matrix operand order repaired and checked by reading, not compiled, not run on hardware
//
#include <hip/hip_runtime.h>
#include <hip/hip_bf16.h>

// CDNA5 / gfx1250 cross-attention, all matmuls via v_wmma_f32_16x16x32_f16.
// Shapes (from reference): b=1 m=4 n=3 c=64 H=W=32, heads=8 dim_head=32.
// Workspace layouts chosen so every WMMA fragment load is contiguous 16B
// (global_load_b128) per lane:
//   Q  f16 [mh][p][d]   (d contiguous)  -> A-frag: 2 x b128
//   K  f16 [mh][j][d]   (d contiguous)  -> QK^T B-frag: 2 x b128
//   V^T f16 [mh][d][j]  (j contiguous)  -> PV  B-frag: 2 x b128
//   AO f16 [m][p][o]    (o contiguous)  -> out-proj A-frag: 2 x b128

typedef __attribute__((ext_vector_type(16))) _Float16      v16h;
typedef __attribute__((ext_vector_type(8)))  float         v8f;
typedef __attribute__((ext_vector_type(4)))  unsigned int  v4u;
typedef __attribute__((ext_vector_type(4)))  float         v4f;

union AFrag { v16h v; _Float16 h[16]; v4u u[2]; };
union CFrag { v8f  v; float    f[8];  };

#define HEADS     8
#define DHEAD     32
#define HW        1024
#define CIN       64
#define MB        4
#define NB        3
#define DIM_INNER 256
#define JTOT      3072   // n * H * W

#define WMMA_F16(A, B, C) \
  __builtin_amdgcn_wmma_f32_16x16x32_f16(false, (A), false, (B), (short)0, (C), false, false)

// A-fragment (16x32 f16): element e -> K = half*16 + e
__device__ __forceinline__ int a_k(int e, int half) {
  return half * 16 + e;
}

// Load 16 consecutive fp32 from wrow, convert to the 16 f16 elements of a
// B-fragment lane (B element e -> K = half*16 + e, already folded into wrow).
__device__ __forceinline__ void load_w16(const float* __restrict__ wrow, AFrag& b) {
  const v4f* w4 = (const v4f*)wrow;
#pragma unroll
  for (int i = 0; i < 4; ++i) {
    v4f w = w4[i];
    b.h[4 * i + 0] = (_Float16)w.x;
    b.h[4 * i + 1] = (_Float16)w.y;
    b.h[4 * i + 2] = (_Float16)w.z;
    b.h[4 * i + 3] = (_Float16)w.w;
  }
}

// ---------------- Stage 1: Q projection (+scale), store f16 [mh][p][d] ------
__global__ __launch_bounds__(32)
void ca_proj_q(const float* __restrict__ x1, const float* __restrict__ Wq,
               _Float16* __restrict__ qws) {
  const int pt = blockIdx.x;          // 0..63 query-row tile
  const int m  = blockIdx.y;          // 0..3
  const int lane = threadIdx.x, lrow = lane & 15, half = lane >> 4;
  const int p0 = pt * 16;
  const float scale = 0.17677669529663687f;  // 32^-0.5

  AFrag a0, a1;
  const float* xb = x1 + (size_t)m * CIN * HW;
#pragma unroll
  for (int e = 0; e < 16; ++e) {      // channel-major input: strided scalars
    int k = a_k(e, half);
    a0.h[e] = (_Float16)xb[(size_t)k * HW + p0 + lrow];
    a1.h[e] = (_Float16)xb[(size_t)(k + 32) * HW + p0 + lrow];
  }

  for (int ot = 0; ot < 16; ++ot) {   // 256 output channels / 16
    const int o = ot * 16 + lrow;
    AFrag b0, b1;
    load_w16(Wq + (size_t)o * CIN + half * 16, b0);        // K = 0..31
    load_w16(Wq + (size_t)o * CIN + 32 + half * 16, b1);   // K = 32..63
    v8f z = {};
    CFrag acc;
    acc.v = WMMA_F16(a0.v, b0.v, z);
    acc.v = WMMA_F16(a1.v, b1.v, acc.v);

    const int h = o >> 5, d = o & 31, mh = m * HEADS + h;
#pragma unroll
    for (int r = 0; r < 8; ++r) {
      int p = p0 + r + half * 8;
      qws[((size_t)mh * HW + p) * DHEAD + d] = (_Float16)(acc.f[r] * scale);
    }
  }
}

// ------- Stage 2: K/V projection; K -> [mh][j][d], V -> transposed [mh][d][j]
__global__ __launch_bounds__(32)
void ca_proj_kv(const float* __restrict__ x2, const float* __restrict__ Wkv,
                _Float16* __restrict__ kws, _Float16* __restrict__ vT) {
  const int pt = blockIdx.x;          // 0..63
  const int n  = blockIdx.y;          // 0..2
  const int m  = blockIdx.z;          // 0..3
  const int lane = threadIdx.x, lrow = lane & 15, half = lane >> 4;
  const int p0 = pt * 16;

  AFrag a0, a1;
  const float* xb = x2 + (size_t)(m * NB + n) * CIN * HW;
#pragma unroll
  for (int e = 0; e < 16; ++e) {
    int k = a_k(e, half);
    a0.h[e] = (_Float16)xb[(size_t)k * HW + p0 + lrow];
    a1.h[e] = (_Float16)xb[(size_t)(k + 32) * HW + p0 + lrow];
  }

  for (int ot = 0; ot < 32; ++ot) {   // 512 output channels / 16
    const int o = ot * 16 + lrow;
    AFrag b0, b1;
    load_w16(Wkv + (size_t)o * CIN + half * 16, b0);
    load_w16(Wkv + (size_t)o * CIN + 32 + half * 16, b1);
    v8f z = {};
    CFrag acc;
    acc.v = WMMA_F16(a0.v, b0.v, z);
    acc.v = WMMA_F16(a1.v, b1.v, acc.v);

    if (ot < 16) {                    // K half -> [mh][j][d] (d contiguous)
      const int h = o >> 5, d = o & 31, mh = m * HEADS + h;
#pragma unroll
      for (int r = 0; r < 8; ++r) {
        int p = p0 + r + half * 8;
        int j = n * HW + p;
        kws[((size_t)mh * JTOT + j) * DHEAD + d] = (_Float16)acc.f[r];
      }
    } else {                          // V half -> transposed [mh][d][j]
      const int o2 = o - DIM_INNER;
      const int h = o2 >> 5, d = o2 & 31, mh = m * HEADS + h;
#pragma unroll
      for (int r = 0; r < 8; ++r) {
        int p = p0 + r + half * 8;
        int j = n * HW + p;
        vT[((size_t)mh * DHEAD + d) * JTOT + j] = (_Float16)acc.f[r];
      }
    }
  }
}

// -------- Stage 3: flash attention, one wave per (mh, 16-query tile) --------
__global__ __launch_bounds__(32)
void ca_attn(const _Float16* __restrict__ qws, const _Float16* __restrict__ kws,
             const _Float16* __restrict__ vT, _Float16* __restrict__ aows) {
  __shared__ __align__(16) _Float16 plds[16][32];  // P transpose bounce
  const int mh = blockIdx.x, m = mh >> 3, h = mh & 7;
  const int p0 = blockIdx.y * 16;
  const int lane = threadIdx.x, lrow = lane & 15, half = lane >> 4;

  // Q A-fragment: two contiguous 16B chunks per lane
  AFrag qa;
  const _Float16* qrow = qws + ((size_t)mh * HW + p0 + lrow) * DHEAD;
  qa.u[0] = *(const v4u*)(qrow + half * 16);
  qa.u[1] = *(const v4u*)(qrow + half * 16 + 8);

  const _Float16* kb = kws + (size_t)mh * JTOT * DHEAD;   // [j][d]
  const _Float16* vb = vT + (size_t)mh * DHEAD * JTOT;    // [d][j]

  float rm[8], rl[8];
  CFrag O0, O1;
#pragma unroll
  for (int r = 0; r < 8; ++r) { rm[r] = -1e30f; rl[r] = 0.f; O0.f[r] = 0.f; O1.f[r] = 0.f; }

  for (int jb = 0; jb < JTOT; jb += 32) {
    // QK^T B-fragments: lane reads K[jb+n][half*16 .. half*16+15] (32B)
    AFrag kf0, kf1;
    {
      const _Float16* kr0 = kb + (size_t)(jb + lrow) * DHEAD + half * 16;
      const _Float16* kr1 = kb + (size_t)(jb + 16 + lrow) * DHEAD + half * 16;
      kf0.u[0] = ((const v4u*)kr0)[0];
      kf0.u[1] = ((const v4u*)kr0)[1];
      kf1.u[0] = ((const v4u*)kr1)[0];
      kf1.u[1] = ((const v4u*)kr1)[1];
    }
    if (jb + 32 < JTOT)  // each lane prefetches one of the next 32 K rows
      __builtin_prefetch(kb + (size_t)(jb + 32 + lane) * DHEAD, 0, 0);

    v8f z = {};
    CFrag S0, S1;
    S0.v = WMMA_F16(qa.v, kf0.v, z);
    S1.v = WMMA_F16(qa.v, kf1.v, z);

    // online softmax: row m=r(+8*half) lives in fixed slots across 16 lanes
    float cm[8];
#pragma unroll
    for (int r = 0; r < 8; ++r) cm[r] = fmaxf(S0.f[r], S1.f[r]);
#pragma unroll
    for (int off = 1; off < 16; off <<= 1)
#pragma unroll
      for (int r = 0; r < 8; ++r) cm[r] = fmaxf(cm[r], __shfl_xor(cm[r], off, 32));

    float al[8], cs[8];
#pragma unroll
    for (int r = 0; r < 8; ++r) {
      float mn = fmaxf(rm[r], cm[r]);
      al[r] = __expf(rm[r] - mn);
      rm[r] = mn;
      S0.f[r] = __expf(S0.f[r] - mn);
      S1.f[r] = __expf(S1.f[r] - mn);
      cs[r] = S0.f[r] + S1.f[r];
    }
#pragma unroll
    for (int off = 1; off < 16; off <<= 1)
#pragma unroll
      for (int r = 0; r < 8; ++r) cs[r] += __shfl_xor(cs[r], off, 32);
#pragma unroll
    for (int r = 0; r < 8; ++r) {
      rl[r] = rl[r] * al[r] + cs[r];
      O0.f[r] *= al[r];
      O1.f[r] *= al[r];
    }

    // transpose P (D-layout) -> A-layout via LDS
    __syncthreads();
#pragma unroll
    for (int r = 0; r < 8; ++r) {
      plds[r + half * 8][lrow]      = (_Float16)S0.f[r];
      plds[r + half * 8][16 + lrow] = (_Float16)S1.f[r];
    }
    __syncthreads();
    AFrag pa;
    pa.u[0] = *(const v4u*)(&plds[lrow][half * 16]);
    pa.u[1] = *(const v4u*)(&plds[lrow][half * 16 + 8]);

    // PV B-fragments: lane reads V^T[d][jb + half*16 .. +15] (32B)
    AFrag vf0, vf1;
    {
      const _Float16* vr0 = vb + (size_t)lrow * JTOT + jb + half * 16;
      const _Float16* vr1 = vb + (size_t)(16 + lrow) * JTOT + jb + half * 16;
      vf0.u[0] = ((const v4u*)vr0)[0];
      vf0.u[1] = ((const v4u*)vr0)[1];
      vf1.u[0] = ((const v4u*)vr1)[0];
      vf1.u[1] = ((const v4u*)vr1)[1];
    }
    O0.v = WMMA_F16(pa.v, vf0.v, O0.v);
    O1.v = WMMA_F16(pa.v, vf1.v, O1.v);
  }

  _Float16* ao = aows + (size_t)m * HW * DIM_INNER;
#pragma unroll
  for (int r = 0; r < 8; ++r) {
    int p = p0 + r + half * 8;
    float inv = 1.0f / rl[r];
    ao[(size_t)p * DIM_INNER + h * DHEAD + lrow]      = (_Float16)(O0.f[r] * inv);
    ao[(size_t)p * DIM_INNER + h * DHEAD + 16 + lrow] = (_Float16)(O1.f[r] * inv);
  }
}

// ---------------- Stage 4: output projection [1024x256]x[256x64] ------------
__global__ __launch_bounds__(32)
void ca_proj_out(const _Float16* __restrict__ aows, const float* __restrict__ Wout,
                 float* __restrict__ out) {
  const int pt = blockIdx.x;          // 0..63
  const int ct = blockIdx.y;          // 0..3
  const int m  = blockIdx.z;          // 0..3
  const int lane = threadIdx.x, lrow = lane & 15, half = lane >> 4;
  const int p0 = pt * 16, c0 = ct * 16;

  const _Float16* arow = aows + ((size_t)m * HW + p0 + lrow) * DIM_INNER;
  const float*    wrow = Wout + (size_t)(c0 + lrow) * DIM_INNER;

  CFrag acc;
  v8f z = {};
  acc.v = z;
  for (int kt = 0; kt < 8; ++kt) {    // K = 256 / 32
    AFrag a, b;
    a.u[0] = *(const v4u*)(arow + kt * 32 + half * 16);
    a.u[1] = *(const v4u*)(arow + kt * 32 + half * 16 + 8);
    load_w16(wrow + kt * 32 + half * 16, b);
    acc.v = WMMA_F16(a.v, b.v, acc.v);
  }
#pragma unroll
  for (int r = 0; r < 8; ++r) {
    int p = p0 + r + half * 8;
    int c = c0 + lrow;
    out[((size_t)m * CIN + c) * HW + p] = acc.f[r];
  }
}

extern "C" void kernel_launch(void* const* d_in, const int* in_sizes, int n_in,
                              void* d_out, int out_size, void* d_ws, size_t ws_size,
                              hipStream_t stream) {
  const float* x1   = (const float*)d_in[0];
  const float* x2   = (const float*)d_in[1];
  const float* Wq   = (const float*)d_in[2];
  const float* Wkv  = (const float*)d_in[3];
  const float* Wout = (const float*)d_in[4];
  float* out = (float*)d_out;

  // Workspace (16 MB total):
  //   [ 0, 2MB) Q    f16 [32][1024][32]
  //   [ 2, 8MB) K    f16 [32][3072][32]
  //   [ 8,14MB) V^T  f16 [32][32][3072]
  //   [14,16MB) attn f16 [4][1024][256]
  char* ws = (char*)d_ws;
  _Float16* qws = (_Float16*)(ws);
  _Float16* kws = (_Float16*)(ws + (size_t)(2u << 20));
  _Float16* vT  = (_Float16*)(ws + (size_t)(8u << 20));
  _Float16* ao  = (_Float16*)(ws + (size_t)(14u << 20));

  ca_proj_q  <<<dim3(64, 4, 1), 32, 0, stream>>>(x1, Wq, qws);
  ca_proj_kv <<<dim3(64, 3, 4), 32, 0, stream>>>(x2, Wkv, kws, vT);
  ca_attn    <<<dim3(32, 64, 1), 32, 0, stream>>>(qws, kws, vT, ao);
  ca_proj_out<<<dim3(64, 4, 4), 32, 0, stream>>>(ao, Wout, out);
}
